// SymGatedGCNMambaModel_45844480918237
// MI455X (gfx1250) — compile-verified, matrix operand order checked by reading, not run on hardware
//
#include <hip/hip_runtime.h>

// ---------------- model constants ----------------
#define NN 10000
#define EE 160000
#define LL 256
#define HH 64
#define II 16
#define NLAYERS 8
#define DM 4
#define DS 32
#define DC 4
#define DI 8
#define DTR 1
#define SS 64
#define EPS 1e-6f

typedef __attribute__((ext_vector_type(16))) __bf16 v16bf;
typedef __attribute__((ext_vector_type(8)))  float  v8f;

// fp32 -> bf16 bits, round-to-nearest-even
static __device__ __forceinline__ unsigned short f2bfbits(float f) {
  unsigned int u = __builtin_bit_cast(unsigned int, f);
  u += 0x7FFFu + ((u >> 16) & 1u);
  return (unsigned short)(u >> 16);
}
static __device__ __forceinline__ float sigmoidf_(float x) {
  return 1.0f / (1.0f + __expf(-x));
}

// Inverse of the CDNA5 16-bit WMMA operand layout (16x32 A tile / 32x16 B tile):
// for within-slab K (0..31) and row/col r (0..15):
//   lane = r + 16 * K[3];  elem = (K & 7) + 8 * (K >= 16)
static __device__ __forceinline__ int frag_lane(int r, int kr) {
  return r + (((kr >> 3) & 1) << 4);
}
static __device__ __forceinline__ int frag_elem(int kr) {
  return (kr & 7) + ((kr >= 16) ? 8 : 0);
}

// ---------------- small encoder MLP: out = relu(X@W1+b1)@W2+b2 ----------------
__global__ void encoder_kernel(const float* __restrict__ X,
                               const float* __restrict__ W1, const float* __restrict__ b1,
                               const float* __restrict__ W2, const float* __restrict__ b2,
                               float* __restrict__ out, int M) {
  __shared__ float hid[4][II];
  int tid = threadIdx.x;
  int lr  = tid >> 6;
  int c   = tid & 63;
  int row = blockIdx.x * 4 + lr;
  if (c < II) {
    float s = b1[c];
    if (row < M) {
      s += X[row * 2 + 0] * W1[0 * II + c];
      s += X[row * 2 + 1] * W1[1 * II + c];
    }
    hid[lr][c] = s > 0.f ? s : 0.f;
  }
  __syncthreads();
  if (row < M) {
    float s = b2[c];
#pragma unroll
    for (int j = 0; j < II; ++j) s += hid[lr][j] * W2[j * HH + c];
    out[row * HH + c] = s;
  }
}

// ---------------- WMMA GEMM: out = A(Mx64) @ W(64x64) + bias ----------------
// Block = 256 threads (8 waves), 64 rows x 64 cols per block.
// Tiles staged in LDS in fragment order -> each fragment is 2x ds_load_b128.
__global__ void gemm64_wmma(const float* __restrict__ A, const float* __restrict__ W,
                            const float* __restrict__ bias, float* __restrict__ out, int M) {
  __shared__ alignas(32) unsigned short aF[2][4][32][16];  // [kslab][rowgrp][lane][elem] 8 KB
  __shared__ alignas(32) unsigned short wF[2][4][32][16];  // [kslab][coltile][lane][elem] 8 KB
  int tid = threadIdx.x;
  int r0  = blockIdx.x * 64;

  // stage W (64x64) fragment-ordered
  for (int q = tid; q < HH * HH; q += 256) {
    int K = q >> 6, c = q & 63;
    int kr = K & 31;
    wF[K >> 5][c >> 4][frag_lane(c & 15, kr)][frag_elem(kr)] = f2bfbits(W[q]);
  }
  // stage A (64 rows x 64 K) fragment-ordered, coalesced global reads
  for (int q = tid; q < 64 * HH; q += 256) {
    int r = q >> 6, c = q & 63;
    int row = r0 + r;
    float v = (row < M) ? A[(size_t)row * HH + c] : 0.f;
    int kr = c & 31;
    aF[c >> 5][r >> 4][frag_lane(r & 15, kr)][frag_elem(kr)] = f2bfbits(v);
  }
  __syncthreads();

  int wave = tid >> 5, lane = tid & 31;
  int g0 = wave >> 2, ct = wave & 3;   // row-groups g0 and g0+2, col-tile ct

  v8f acc0 = {}, acc1 = {};
#pragma unroll
  for (int s = 0; s < 2; ++s) {
    v16bf b  = *(const v16bf*)&wF[s][ct][lane][0];
    v16bf a0 = *(const v16bf*)&aF[s][g0][lane][0];
    v16bf a1 = *(const v16bf*)&aF[s][g0 + 2][lane][0];
    acc0 = __builtin_amdgcn_wmma_f32_16x16x32_bf16(false, a0, false, b, (short)0, acc0,
                                                   false, false);
    acc1 = __builtin_amdgcn_wmma_f32_16x16x32_bf16(false, a1, false, b, (short)0, acc1,
                                                   false, false);
  }

  int col  = ct * 16 + (lane & 15);
  float bb = bias[col];
  int half = (lane < 16) ? 0 : 8;
  int rowA = r0 + g0 * 16 + half;          // 8 contiguous rows per accumulator
  int rowB = r0 + (g0 + 2) * 16 + half;
  if (r0 + 64 <= M) {                      // fast path: no per-row guards
#pragma unroll
    for (int j = 0; j < 8; ++j) out[(size_t)(rowA + j) * HH + col] = acc0[j] + bb;
#pragma unroll
    for (int j = 0; j < 8; ++j) out[(size_t)(rowB + j) * HH + col] = acc1[j] + bb;
  } else {
#pragma unroll
    for (int j = 0; j < 8; ++j)
      if (rowA + j < M) out[(size_t)(rowA + j) * HH + col] = acc0[j] + bb;
#pragma unroll
    for (int j = 0; j < 8; ++j)
      if (rowB + j < M) out[(size_t)(rowB + j) * HH + col] = acc1[j] + bb;
  }
}

// ---------------- zero ----------------
__global__ void zero_kernel(float* __restrict__ p, int n) {
  int i = blockIdx.x * 256 + threadIdx.x;
  if (i < n) p[i] = 0.f;
}

// ---------------- GNN edge update + segment scatter ----------------
__global__ void gnn_edge_scatter(const float* __restrict__ B1h, const float* __restrict__ B2h,
                                 const float* __restrict__ B3e, const float* __restrict__ A2h,
                                 const float* __restrict__ A3h, const int* __restrict__ src,
                                 const int* __restrict__ dst, float* __restrict__ e_feat,
                                 float* __restrict__ hf_num, float* __restrict__ hf_den,
                                 float* __restrict__ hb_num, float* __restrict__ hb_den,
                                 int nelem) {
  int idx = blockIdx.x * 256 + threadIdx.x;
  if (idx >= nelem) return;
  int i = idx >> 6, c = idx & 63;
  int s = src[i], d = dst[i];
  float raw = B1h[s * HH + c] + B2h[d * HH + c] + B3e[idx];
  float ev = e_feat[idx] + (raw > 0.f ? raw : 0.f);
  e_feat[idx] = ev;
  float sig = sigmoidf_(ev);
  atomicAdd(&hf_num[d * HH + c], A2h[s * HH + c] * sig);
  atomicAdd(&hf_den[d * HH + c], sig);
  atomicAdd(&hb_num[s * HH + c], A3h[d * HH + c] * sig);
  atomicAdd(&hb_den[s * HH + c], sig);
}

// ---------------- GNN node update ----------------
__global__ void gnn_node_update(float* __restrict__ h, const float* __restrict__ A1h,
                                const float* __restrict__ hf_num, const float* __restrict__ hf_den,
                                const float* __restrict__ hb_num, const float* __restrict__ hb_den,
                                int nelem) {
  int idx = blockIdx.x * 256 + threadIdx.x;
  if (idx >= nelem) return;
  float v = A1h[idx] + hf_num[idx] / (hf_den[idx] + EPS) + hb_num[idx] / (hb_den[idx] + EPS);
  h[idx] += (v > 0.f ? v : 0.f);
}

// ---------------- fused Mamba: one block (256 threads) per sequence ----------------
__global__ void mamba_kernel(const float* __restrict__ reads, const int* __restrict__ read_length,
                             const float* __restrict__ in_proj_W, const float* __restrict__ conv_W,
                             const float* __restrict__ conv_b, const float* __restrict__ x_proj_W,
                             const float* __restrict__ dt_proj_W, const float* __restrict__ dt_proj_b,
                             const float* __restrict__ A_log, const float* __restrict__ Dskip,
                             const float* __restrict__ out_proj_W, const float* __restrict__ base_W,
                             const float* __restrict__ base_b, float* __restrict__ h) {
  constexpr int T = 64;                 // time chunk
  __shared__ float sh_xh[LL][DI];       // 8 KB : pre-conv x
  __shared__ float sh_z[LL][DI];        // 8 KB : gate
  __shared__ float sh_dt[T][DI];        // 2 KB
  __shared__ float sh_B[T][DS];         // 8 KB
  __shared__ float sh_C[T][DS];         // 8 KB
  __shared__ float sh_xc[T][DI];        // 2 KB : post-conv silu(x)
  __shared__ float sh_yt[DI];
  __shared__ float sh_x2[DM];

  int n = blockIdx.x;
  int tid = threadIdx.x;

  // phase 1: in-projection, t == tid
  {
    float r[DM];
#pragma unroll
    for (int m = 0; m < DM; ++m) r[m] = reads[((size_t)n * LL + tid) * DM + m];
#pragma unroll
    for (int j = 0; j < 2 * DI; ++j) {
      float s = 0.f;
#pragma unroll
      for (int m = 0; m < DM; ++m) s += r[m] * in_proj_W[m * (2 * DI) + j];
      if (j < DI) sh_xh[tid][j] = s;
      else        sh_z[tid][j - DI] = s;
    }
  }
  int target = read_length[n] - 1;
  int d = tid >> 5;       // wave id == channel
  int s = tid & 31;       // lane == state index
  float Aval = -__expf(A_log[d * DS + s]);
  float hstate = 0.f;
  __syncthreads();

  for (int ch = 0; ch < LL / T; ++ch) {
    if (tid < T) {
      int t = ch * T + tid;
      float xcv[DI];
#pragma unroll
      for (int dd = 0; dd < DI; ++dd) {
        float acc = conv_b[dd];
#pragma unroll
        for (int k = 0; k < DC; ++k) {
          int tt = t + k - (DC - 1);
          if (tt >= 0) acc += sh_xh[tt][dd] * conv_W[dd * DC + k];
        }
        float act = acc * sigmoidf_(acc);   // silu
        xcv[dd] = act;
        sh_xc[tid][dd] = act;
      }
      // dbl = xcv @ x_proj_W (8 x 65); j=0 -> dt, 1..32 -> B, 33..64 -> C
      for (int j = 0; j < DTR + 2 * DS; ++j) {
        float acc = 0.f;
#pragma unroll
        for (int dd = 0; dd < DI; ++dd) acc += xcv[dd] * x_proj_W[dd * (DTR + 2 * DS) + j];
        if (j == 0) {
#pragma unroll
          for (int dd = 0; dd < DI; ++dd) {
            float raw = acc * dt_proj_W[dd] + dt_proj_b[dd];
            sh_dt[tid][dd] = (raw > 20.f) ? raw : log1pf(__expf(raw));  // softplus
          }
        } else if (j < 1 + DS) sh_B[tid][j - 1] = acc;
        else                   sh_C[tid][j - 1 - DS] = acc;
      }
    }
    __syncthreads();
    // selective scan: wave d owns channel d, lane s owns state s (wave32)
    for (int t = 0; t < T; ++t) {
      float dt = sh_dt[t][d];
      hstate = __expf(dt * Aval) * hstate + dt * sh_B[t][s] * sh_xc[t][d];
      float y = hstate * sh_C[t][s];
#pragma unroll
      for (int o = 16; o > 0; o >>= 1) y += __shfl_xor(y, o, 32);
      int tg = ch * T + t;
      if (s == 0 && tg == target) {
        float yf = y + sh_xc[t][d] * Dskip[d];
        float zz = sh_z[tg][d];
        yf *= zz * sigmoidf_(zz);          // * silu(z)
        sh_yt[d] = yf;
      }
    }
    __syncthreads();
  }

  if (tid < DM) {
    float acc = 0.f;
#pragma unroll
    for (int dd = 0; dd < DI; ++dd) acc += sh_yt[dd] * out_proj_W[dd * DM + tid];
    sh_x2[tid] = acc;
  }
  __syncthreads();
  if (tid < HH) {
    float acc = base_b[tid];
#pragma unroll
    for (int m = 0; m < DM; ++m) acc += sh_x2[m] * base_W[m * HH + tid];
    h[(size_t)n * HH + tid] += acc;
  }
}

// ---------------- fused final MLP (WMMA): relu([h_src,h_dst,e]@Wp1+bp1)@Wp2+bp2 ----------------
// Block = 256 threads, 32 edges per block. K = 192 -> 6 wmma steps, fragment-ordered LDS.
__global__ void final_mlp_wmma(const float* __restrict__ h, const float* __restrict__ e_feat,
                               const int* __restrict__ src, const int* __restrict__ dst,
                               const float* __restrict__ Wp1, const float* __restrict__ bp1,
                               const float* __restrict__ Wp2, const float* __restrict__ bp2,
                               float* __restrict__ out, int Etot) {
  __shared__ alignas(32) unsigned short wF[6][4][32][16];  // 24 KB
  __shared__ alignas(32) unsigned short aF[6][2][32][16];  // 12 KB
  __shared__ float sAct[32][SS];                           // 8 KB
  int tid = threadIdx.x;
  int e0 = blockIdx.x * 32;

  // stage Wp1 (192x64) fragment-ordered
  for (int q = tid; q < 3 * HH * SS; q += 256) {
    int K = q >> 6, c = q & 63;
    int kr = K & 31;
    wF[K >> 5][c >> 4][frag_lane(c & 15, kr)][frag_elem(kr)] = f2bfbits(Wp1[q]);
  }
  // stage gathered cat-rows (32 x 192) fragment-ordered
  for (int q = tid; q < 32 * 3 * HH; q += 256) {
    int r = q / (3 * HH), c = q % (3 * HH);
    int i = e0 + r;
    float v = 0.f;
    if (i < Etot) {
      if (c < HH)           v = h[src[i] * HH + c];
      else if (c < 2 * HH)  v = h[dst[i] * HH + (c - HH)];
      else                  v = e_feat[(size_t)i * HH + (c - 2 * HH)];
    }
    int kr = c & 31;
    aF[c >> 5][r >> 4][frag_lane(r & 15, kr)][frag_elem(kr)] = f2bfbits(v);
  }
  __syncthreads();

  int wave = tid >> 5, lane = tid & 31;
  int g = wave >> 2, ct = wave & 3;

  v8f acc = {};
#pragma unroll
  for (int s = 0; s < 6; ++s) {
    v16bf a = *(const v16bf*)&aF[s][g][lane][0];
    v16bf b = *(const v16bf*)&wF[s][ct][lane][0];
    acc = __builtin_amdgcn_wmma_f32_16x16x32_bf16(false, a, false, b, (short)0, acc,
                                                  false, false);
  }
  int col = ct * 16 + (lane & 15);
  float bb = bp1[col];
#pragma unroll
  for (int j = 0; j < 8; ++j) {
    int m = g * 16 + j + ((lane < 16) ? 0 : 8);
    float v = acc[j] + bb;
    sAct[m][col] = v > 0.f ? v : 0.f;
  }
  __syncthreads();

  // second layer: 256 threads, 8 lanes per edge row, butterfly reduce
  {
    int r = tid >> 3;          // edge row 0..31
    int part = tid & 7;        // 8-col strip
    float partial = 0.f;
#pragma unroll
    for (int c = 0; c < 8; ++c) {
      int cc = part * 8 + c;
      partial += sAct[r][cc] * Wp2[cc];
    }
#pragma unroll
    for (int o = 4; o > 0; o >>= 1) partial += __shfl_xor(partial, o, 32);
    int i = e0 + r;
    if (part == 0 && i < Etot) out[i] = partial + bp2[0];
  }
}

// ---------------- host-side orchestration ----------------
extern "C" void kernel_launch(void* const* d_in, const int* in_sizes, int n_in,
                              void* d_out, int out_size, void* d_ws, size_t ws_size,
                              hipStream_t stream) {
  const float* x        = (const float*)d_in[0];
  const float* e_in     = (const float*)d_in[1];
  const float* reads    = (const float*)d_in[2];
  const int*   src      = (const int*)d_in[3];
  const int*   dst      = (const int*)d_in[4];
  const int*   rlen     = (const int*)d_in[5];
  const float* W1n      = (const float*)d_in[6];
  const float* b1n      = (const float*)d_in[7];
  const float* W2n      = (const float*)d_in[8];
  const float* b2n      = (const float*)d_in[9];
  const float* W1e      = (const float*)d_in[10];
  const float* b1e      = (const float*)d_in[11];
  const float* W2e      = (const float*)d_in[12];
  const float* b2e      = (const float*)d_in[13];
  const float* gnn_W    = (const float*)d_in[14];
  const float* gnn_b    = (const float*)d_in[15];
  const float* in_proj_W  = (const float*)d_in[16];
  const float* conv_W   = (const float*)d_in[17];
  const float* conv_b   = (const float*)d_in[18];
  const float* x_proj_W = (const float*)d_in[19];
  const float* dt_proj_W  = (const float*)d_in[20];
  const float* dt_proj_b  = (const float*)d_in[21];
  const float* A_log    = (const float*)d_in[22];
  const float* Dskip    = (const float*)d_in[23];
  const float* out_proj_W = (const float*)d_in[24];
  const float* base_W   = (const float*)d_in[25];
  const float* base_b   = (const float*)d_in[26];
  const float* Wp1      = (const float*)d_in[27];
  const float* bp1      = (const float*)d_in[28];
  const float* Wp2      = (const float*)d_in[29];
  const float* bp2      = (const float*)d_in[30];
  float* out = (float*)d_out;

  const int N = in_sizes[5];    // read_length count
  const int E = in_sizes[3];    // src count
  const size_t NH = (size_t)N * HH;
  const size_t EH = (size_t)E * HH;

  float* ws = (float*)d_ws;
  size_t off = 0;
  float* h      = ws + off; off += NH;
  float* A1h    = ws + off; off += NH;
  float* A2h    = ws + off; off += NH;
  float* A3h    = ws + off; off += NH;
  float* B1h    = ws + off; off += NH;
  float* B2h    = ws + off; off += NH;
  float* segbuf = ws + off; off += 4 * NH;   // hf_num, hf_den, hb_num, hb_den
  float* hf_num = segbuf;
  float* hf_den = segbuf + NH;
  float* hb_num = segbuf + 2 * NH;
  float* hb_den = segbuf + 3 * NH;
  float* e_feat = ws + off; off += EH;
  float* B3e    = ws + off; off += EH;

  const dim3 blk(256);

  // encoders
  encoder_kernel<<<dim3((N + 3) / 4), blk, 0, stream>>>(x, W1n, b1n, W2n, b2n, h, N);
  encoder_kernel<<<dim3((E + 3) / 4), blk, 0, stream>>>(e_in, W1e, b1e, W2e, b2e, e_feat, E);

  const int gN = (N + 63) / 64;
  const int gE = (E + 63) / 64;

  // GNN layers
  for (int l = 0; l < NLAYERS; ++l) {
    const float* Wl = gnn_W + (size_t)l * 6 * HH * HH;
    const float* bl = gnn_b + (size_t)l * 6 * HH;
    gemm64_wmma<<<dim3(gN), blk, 0, stream>>>(h, Wl + 0 * HH * HH, bl + 0 * HH, A1h, N);
    gemm64_wmma<<<dim3(gN), blk, 0, stream>>>(h, Wl + 1 * HH * HH, bl + 1 * HH, A2h, N);
    gemm64_wmma<<<dim3(gN), blk, 0, stream>>>(h, Wl + 2 * HH * HH, bl + 2 * HH, A3h, N);
    gemm64_wmma<<<dim3(gN), blk, 0, stream>>>(h, Wl + 3 * HH * HH, bl + 3 * HH, B1h, N);
    gemm64_wmma<<<dim3(gN), blk, 0, stream>>>(h, Wl + 4 * HH * HH, bl + 4 * HH, B2h, N);
    gemm64_wmma<<<dim3(gE), blk, 0, stream>>>(e_feat, Wl + 5 * HH * HH, bl + 5 * HH, B3e, E);

    int segN = (int)(4 * NH);
    zero_kernel<<<dim3((segN + 255) / 256), blk, 0, stream>>>(segbuf, segN);

    int nelemE = E * HH;
    gnn_edge_scatter<<<dim3((nelemE + 255) / 256), blk, 0, stream>>>(
        B1h, B2h, B3e, A2h, A3h, src, dst, e_feat, hf_num, hf_den, hb_num, hb_den, nelemE);

    int nelemN = N * HH;
    gnn_node_update<<<dim3((nelemN + 255) / 256), blk, 0, stream>>>(
        h, A1h, hf_num, hf_den, hb_num, hb_den, nelemN);
  }

  // fused Mamba (adds base projection of selected timestep into h)
  mamba_kernel<<<dim3(N), blk, 0, stream>>>(reads, rlen, in_proj_W, conv_W, conv_b,
                                            x_proj_W, dt_proj_W, dt_proj_b, A_log, Dskip,
                                            out_proj_W, base_W, base_b, h);

  // final edge predictor
  final_mlp_wmma<<<dim3((E + 31) / 32), blk, 0, stream>>>(h, e_feat, src, dst,
                                                          Wp1, bp1, Wp2, bp2, out, E);
}
